// GNN_SAGEConv_homogen_71073118814466
// MI455X (gfx1250) — matrix operand order repaired and checked by reading, not compiled, not run on hardware
//
#include <hip/hip_runtime.h>
#include <hip/hip_bf16.h>

typedef _Float16 f16;
typedef __attribute__((ext_vector_type(16))) _Float16     v16h;
typedef __attribute__((ext_vector_type(8)))  _Float16     v8h;
typedef __attribute__((ext_vector_type(8)))  float        v8f;
typedef __attribute__((ext_vector_type(4)))  unsigned int v4u;
typedef __attribute__((ext_vector_type(8)))  int          v8i;
typedef __attribute__((ext_vector_type(4)))  int          v4i;

static __device__ __forceinline__ v16h cat8(v8h lo, v8h hi) {
    return __builtin_shufflevector(lo, hi, 0,1,2,3,4,5,6,7,8,9,10,11,12,13,14,15);
}

// ---------------------------------------------------------------------------
// TDM: DMA a 128x128 f16 row-major matrix from global memory into LDS.
// D# per CDNA5 ISA ch.8: group0 = {count|flags, lds_addr, global_addr, type},
// group1 = {data_size, tensor dims/strides, tile dims}. 2D tile, no padding,
// no multicast, no iteration. Issued by wave 0 only; EXEC is ignored by TDM.
// This toolchain carries the 6-arg builtin (extra int32x8 group, then cpol).
// ---------------------------------------------------------------------------
static __device__ __forceinline__ void tdm_load_16k_f16(const f16* gsrc, f16* lds_dst) {
    unsigned long long ga = (unsigned long long)(uintptr_t)gsrc;
    unsigned int ldsa = (unsigned int)(uintptr_t)lds_dst;   // low 32 bits = LDS offset
    v4u g0 = { 1u,                                          // count=1 (valid user D#)
               ldsa,                                        // [63:32]  lds_addr
               (unsigned int)(ga & 0xFFFFFFFFu),            // [95:64]  global_addr lo
               (unsigned int)((ga >> 32) & 0x01FFFFFFu)     // [120:96] global_addr hi
                   | (2u << 30) };                          // [127:126] type=2
    v8i g1 = { (int)(1u << 16),      // [17:16] data_size=1 -> 2 bytes
               (int)(128u << 16),    // [63:48]  tensor_dim0 = 128
               (int)(128u << 16),    // [95:80]  tensor_dim1 = 128
               (int)(128u << 16),    // [127:112] tile_dim0  = 128
               128,                  // [143:128] tile_dim1  = 128
               128,                  // [191:160] tensor_dim0_stride = 128
               0, 0 };
    v4i gz4 = { 0, 0, 0, 0 };
    v8i gz8 = { 0, 0, 0, 0, 0, 0, 0, 0 };
    __builtin_amdgcn_tensor_load_to_lds(g0, g1, gz4, gz4, gz8, 0);
}

// ---------------------------------------------------------------------------
// Weight prep: f32 -> f16, all stored [out][in] row-major (Wb transposed so
// the bilinear GEMM Z = X @ Wb uses the same B-operand addressing).
// ---------------------------------------------------------------------------
__global__ __launch_bounds__(256) void prep_weights(
    const float* __restrict__ w0, const float* __restrict__ w1,
    const float* __restrict__ w2, const float* __restrict__ w3,
    const float* __restrict__ w4, const float* __restrict__ wb,
    f16* __restrict__ o)
{
    int idx = blockIdx.x * 256 + threadIdx.x;
    if (idx >= 6 * 16384) return;
    int m = idx >> 14, j = idx & 16383;
    float v;
    switch (m) {
        case 0: v = w0[j]; break;
        case 1: v = w1[j]; break;
        case 2: v = w2[j]; break;
        case 3: v = w3[j]; break;
        case 4: v = w4[j]; break;
        default: { int g = j >> 7, f = j & 127; v = wb[f * 128 + g]; } break; // Wb^T
    }
    o[idx] = (f16)v;
}

__global__ __launch_bounds__(256) void f32_to_f16_k(
    const float* __restrict__ src, f16* __restrict__ dst, size_t count4)
{
    size_t idx = (size_t)blockIdx.x * 256 + threadIdx.x;   // in float4 units
    if (idx >= count4) return;
    float4 v = ((const float4*)src)[idx];
    union { f16 h[4]; uint2 u; } r;
    r.h[0] = (f16)v.x; r.h[1] = (f16)v.y; r.h[2] = (f16)v.z; r.h[3] = (f16)v.w;
    *(uint2*)(dst + idx * 4) = r.u;
}

// ---------------------------------------------------------------------------
// WMMA GEMM: Y[m,n] = sum_k A1[m,k]*W1[n,k] (+ A2[m,k]*W2[n,k]) (+ bias[n])
// M = nrows, K = N = 128.  One wave -> 16 rows x 128 cols (8 16x16 tiles).
// Block = 256 threads = 8 waves = 128 rows. Weights staged in LDS via TDM.
// ---------------------------------------------------------------------------
template <bool DUAL>
__global__ __launch_bounds__(256) void linear_wmma(
    const f16* __restrict__ A1, const f16* __restrict__ W1,
    const f16* __restrict__ A2, const f16* __restrict__ W2,
    const float* __restrict__ bias, float* __restrict__ Y, int nrows)
{
    __shared__ __align__(128) f16 sW[(DUAL ? 2 : 1) * 16384];

    const int lane = threadIdx.x & 31;
    const int wave = threadIdx.x >> 5;
    const int m0   = blockIdx.x * 128 + wave * 16;

    // Stage weight matrices into LDS with the Tensor Data Mover (wave 0 only;
    // TENSORcnt is per-wave, so wave 0 waits, then the workgroup barrier
    // makes the LDS tile visible to all waves).
    if (wave == 0) {
        tdm_load_16k_f16(W1, sW);
        if (DUAL) tdm_load_16k_f16(W2, sW + 16384);
        __builtin_amdgcn_s_wait_tensorcnt(0);
    }
    __syncthreads();

    if (m0 >= nrows) return;                 // wave-uniform exit, EXEC stays full

    const int rsel = lane >> 4;              // 0: lanes 0-15, 1: lanes 16-31
    const int l15  = lane & 15;
    const int rowA = min(m0 + l15, nrows - 1);

    // A fragments (ISA 16-bit A 16x32 layout):
    //  lanes 0-15 : row M=lane,   halves 0-7 = K+0..7,  halves 8-15 = K+16..23
    //  lanes16-31 : row M=lane-16,halves 0-7 = K+8..15, halves 8-15 = K+24..31
    v16h a1[4], a2[4];
#pragma unroll
    for (int kc = 0; kc < 4; ++kc) {
        const f16* p = A1 + (size_t)rowA * 128 + kc * 32 + rsel * 8;
        a1[kc] = cat8(*(const v8h*)p, *(const v8h*)(p + 16));
    }
    if (DUAL) {
#pragma unroll
        for (int kc = 0; kc < 4; ++kc) {
            const f16* p = A2 + (size_t)rowA * 128 + kc * 32 + rsel * 8;
            a2[kc] = cat8(*(const v8h*)p, *(const v8h*)(p + 16));
        }
    }

    const bool fullTile = (m0 + 16 <= nrows);   // wave-uniform

#pragma unroll
    for (int nt = 0; nt < 8; ++nt) {
        const int nn = nt * 16 + l15;

        float bv = bias ? bias[nn] : 0.0f;
        v8f c = {bv, bv, bv, bv, bv, bv, bv, bv};

        // B (16-bit 32x16): lanes 0-15 hold K=0..7,16..23 of col n, lanes 16-31 K=8..15,24..31
#pragma unroll
        for (int kc = 0; kc < 4; ++kc) {
            const f16* pb = sW + nn * 128 + kc * 32 + rsel * 8;
            v16h b = cat8(*(const v8h*)pb, *(const v8h*)(pb + 16));
            c = __builtin_amdgcn_wmma_f32_16x16x32_f16(
                    false, a1[kc], false, b, (short)0, c, false, false);
        }
        if (DUAL) {
#pragma unroll
            for (int kc = 0; kc < 4; ++kc) {
                const f16* pb = sW + 16384 + nn * 128 + kc * 32 + rsel * 8;
                v16h b = cat8(*(const v8h*)pb, *(const v8h*)(pb + 16));
                c = __builtin_amdgcn_wmma_f32_16x16x32_f16(
                        false, a2[kc], false, b, (short)0, c, false, false);
            }
        }

        // C/D layout: VGPR v: lanes 0-15 -> row m0+v, lanes 16-31 -> row m0+v+8
        float* yp = Y + (size_t)(m0 + rsel * 8) * 128 + nn;
        if (fullTile) {
#pragma unroll
            for (int v = 0; v < 8; ++v) yp[(size_t)v * 128] = c[v];
        } else {
#pragma unroll
            for (int v = 0; v < 8; ++v)
                if (m0 + rsel * 8 + v < nrows) yp[(size_t)v * 128] = c[v];
        }
    }
}

// ---------------------------------------------------------------------------
// Edge scatter-add (mean aggregation numerator + counts). 2 edges per block.
// ---------------------------------------------------------------------------
__global__ __launch_bounds__(256) void scatter_add_k(
    const float* __restrict__ X, const int* __restrict__ src,
    const int* __restrict__ dst, float* __restrict__ AGG,
    float* __restrict__ CNT, int E)
{
    int e = blockIdx.x * 2 + (threadIdx.x >> 7);
    if (e >= E) return;
    int f = threadIdx.x & 127;
    int s = src[e], d = dst[e];
    atomicAdd(&AGG[(size_t)d * 128 + f], X[(size_t)s * 128 + f]);
    if (f == 0) atomicAdd(&CNT[d], 1.0f);
}

__global__ __launch_bounds__(256) void div_to_f16_k(
    const float* __restrict__ AGG, const float* __restrict__ CNT,
    f16* __restrict__ AGGh, size_t count4)
{
    size_t idx = (size_t)blockIdx.x * 256 + threadIdx.x;   // in float4 units
    if (idx >= count4) return;
    float rc = __frcp_rn(fmaxf(CNT[(idx * 4) >> 7], 1.0f));
    float4 v = ((const float4*)AGG)[idx];
    union { f16 h[4]; uint2 u; } r;
    r.h[0] = (f16)(v.x * rc); r.h[1] = (f16)(v.y * rc);
    r.h[2] = (f16)(v.z * rc); r.h[3] = (f16)(v.w * rc);
    *(uint2*)(AGGh + idx * 4) = r.u;
}

// ---------------------------------------------------------------------------
// score[e] = dot(Z[e1], X[e2]) + bb   (one wave per edge, 4 floats/lane)
// ---------------------------------------------------------------------------
__global__ __launch_bounds__(256) void edge_score_k(
    const float* __restrict__ Z, const float* __restrict__ X,
    const int* __restrict__ e1, const int* __restrict__ e2,
    const float* __restrict__ bb, float* __restrict__ out, int E)
{
    int w = blockIdx.x * 8 + (threadIdx.x >> 5);
    int lane = threadIdx.x & 31;
    if (w >= E) return;
    size_t i1 = (size_t)e1[w], i2 = (size_t)e2[w];
    float4 a = ((const float4*)(Z + i1 * 128))[lane];
    float4 b = ((const float4*)(X + i2 * 128))[lane];
    float s = a.x * b.x + a.y * b.y + a.z * b.z + a.w * b.w;
#pragma unroll
    for (int off = 16; off > 0; off >>= 1) s += __shfl_xor(s, off, 32);
    if (lane == 0) out[w] = s + bb[0];
}

// ---------------------------------------------------------------------------
extern "C" void kernel_launch(void* const* d_in, const int* in_sizes, int n_in,
                              void* d_out, int out_size, void* d_ws, size_t ws_size,
                              hipStream_t stream)
{
    const float* x_in  = (const float*)d_in[0];
    const int*   ei_sc = (const int*)d_in[1];   // [2, E_score]
    const int*   ei_ms = (const int*)d_in[2];   // [2, E_msg]
    const float* W_in  = (const float*)d_in[3];
    const float* b_in  = (const float*)d_in[4];
    const float* Wl1   = (const float*)d_in[5];
    const float* bl1   = (const float*)d_in[6];
    const float* Wr1   = (const float*)d_in[7];
    const float* Wl2   = (const float*)d_in[8];
    const float* bl2   = (const float*)d_in[9];
    const float* Wr2   = (const float*)d_in[10];
    const float* Wb    = (const float*)d_in[11];
    const float* bb    = (const float*)d_in[12];
    float* out = (float*)d_out;

    const int n  = in_sizes[0] / 128;
    const int Es = in_sizes[1] / 2;
    const int Em = in_sizes[2] / 2;
    const size_t NB = (size_t)n * 128;

    // workspace carve (256B aligned)
    char* p = (char*)d_ws;
    auto carve = [&](size_t bytes) { char* r = p; p += (bytes + 255) & ~(size_t)255; return r; };
    float* Yf   = (float*)carve(NB * 4);       // current node features (f32)
    float* AGG  = (float*)carve(NB * 4);       // scatter-sum accumulator
    float* Zf   = (float*)carve(NB * 4);       // bilinear-left precompute
    f16*   Ah   = (f16*)  carve(NB * 2);       // f16 copy of current features
    f16*   AGGh = (f16*)  carve(NB * 2);       // f16 mean-aggregated features
    float* CNT  = (float*)carve((size_t)n * 4);
    f16*   Wh   = (f16*)  carve(6 * 16384 * 2);
    f16 *Winh = Wh,           *Wl1h = Wh + 16384, *Wr1h = Wh + 2 * 16384,
        *Wl2h = Wh + 3*16384, *Wr2h = Wh + 4*16384, *WbTh = Wh + 5 * 16384;

    const size_t NB4    = NB / 4;
    const int cvtGrid   = (int)((NB4 + 255) / 256);
    const int gemmGrid  = (n + 127) / 128;

    // 1) weights -> f16 (Wb transposed)
    prep_weights<<<(6 * 16384 + 255) / 256, 256, 0, stream>>>(
        W_in, Wl1, Wr1, Wl2, Wr2, Wb, Wh);

    // 2) x = x_input @ W_in^T + b_in
    f32_to_f16_k<<<cvtGrid, 256, 0, stream>>>(x_in, Ah, NB4);
    linear_wmma<false><<<gemmGrid, 256, 0, stream>>>(
        Ah, Winh, nullptr, nullptr, b_in, Yf, n);

    // 3) two SAGEConv layers
    for (int layer = 0; layer < 2; ++layer) {
        const f16*   Wl = layer ? Wl2h : Wl1h;
        const f16*   Wr = layer ? Wr2h : Wr1h;
        const float* bl = layer ? bl2  : bl1;

        f32_to_f16_k<<<cvtGrid, 256, 0, stream>>>(Yf, Ah, NB4);
        (void)hipMemsetAsync(AGG, 0, NB * 4, stream);
        (void)hipMemsetAsync(CNT, 0, (size_t)n * 4, stream);
        scatter_add_k<<<(Em + 1) / 2, 256, 0, stream>>>(
            Yf, ei_ms, ei_ms + Em, AGG, CNT, Em);
        div_to_f16_k<<<cvtGrid, 256, 0, stream>>>(AGG, CNT, AGGh, NB4);
        linear_wmma<true><<<gemmGrid, 256, 0, stream>>>(
            AGGh, Wl, Ah, Wr, bl, Yf, n);
    }

    // 4) bilinear: Z = X @ Wb  (via WbT stored [out][in]), then per-edge dot
    f32_to_f16_k<<<cvtGrid, 256, 0, stream>>>(Yf, Ah, NB4);
    linear_wmma<false><<<gemmGrid, 256, 0, stream>>>(
        Ah, WbTh, nullptr, nullptr, nullptr, Zf, n);
    edge_score_k<<<(Es + 7) / 8, 256, 0, stream>>>(
        Zf, Yf, ei_sc, ei_sc + Es, bb, out, Es);
}
